// CustomGRU_63067299774555
// MI455X (gfx1250) — matrix operand order repaired and checked by reading, not compiled, not run on hardware
//
#include <hip/hip_runtime.h>
#include <math.h>

// ---------------------------------------------------------------------------
// GRU forward for T=256, B=64, D=H=1024 on gfx1250 (MI455X).
//   - bf16 WMMA (v_wmma_f32_16x16x32_bf16) for all GEMMs, f32 accumulate
//   - persistent kernel runs the 256-step recurrence with a device-wide
//     generation barrier (2 syncs per step)
//   - operands pre-packed into ISA 7.12.2 A/B fragment layouts
//   - A fragments streamed to LDS via GLOBAL_LOAD_ASYNC_TO_LDS_B128 with
//     double buffering (fallback: register-staged clause copy)
// ---------------------------------------------------------------------------

#define T_STEPS 256
#define BATCH   64
#define HID     1024
#define G3      3072            // 3*HID
#define NT      192             // G3 / 16 column tiles
#define NBLK    48              // persistent grid
#define THREADS 256
#define HPK_ELEMS (32 * 4 * 512)   // one ping-pong slot of packed h frags (=B*H)
#define CHUNK   (8 * 4 * 512)      // 8 kb x 4 mt fragments = 16384 ushorts = 32KB

typedef __attribute__((ext_vector_type(16))) __bf16 v16bf;
typedef __attribute__((ext_vector_type(8)))  float  v8f;
typedef __attribute__((__vector_size__(4 * sizeof(int)))) int vi4_t;
typedef __attribute__((address_space(1))) vi4_t as1_vi4;   // global int4
typedef __attribute__((address_space(3))) vi4_t as3_vi4;   // LDS int4

#if defined(__gfx1250__) && __has_builtin(__builtin_amdgcn_global_load_async_to_lds_b128)
#define GRU_ASYNC_LDS 1
#else
#define GRU_ASYNC_LDS 0
#endif

// ---- helpers --------------------------------------------------------------

__device__ __forceinline__ unsigned short f2bf(float f) {
  unsigned u = __builtin_bit_cast(unsigned, f);
  u += 0x7FFFu + ((u >> 16) & 1u);             // round-to-nearest-even
  return (unsigned short)(u >> 16);
}

__device__ __forceinline__ float sigmoid_f(float x) {
  return 1.0f / (1.0f + __expf(-x));
}

__device__ __forceinline__ v8f bf16_mma(v16bf a, v16bf b, v8f c) {
  // D = A(16x32 bf16) x B(32x16 bf16) + C(16x16 f32)
  return __builtin_amdgcn_wmma_f32_16x16x32_bf16(
      /*neg_a=*/false, a, /*neg_b=*/false, b,
      /*c_mod=*/(short)0, c, /*reuse_a=*/false, /*reuse_b=*/false);
}

// packed A-fragment position of element (b, j) of h  (ISA 16-bit A 16x32 layout)
__device__ __forceinline__ int hpk_index(int b, int j) {
  int mt = b >> 4, m = b & 15;
  int kbl = j >> 5, ko = j & 31;
  int lane, jj;
  if (ko < 8)       { lane = m;      jj = ko;      }
  else if (ko < 16) { lane = m + 16; jj = ko - 8;  }
  else if (ko < 24) { lane = m;      jj = ko - 8;  }
  else              { lane = m + 16; jj = ko - 16; }
  return (kbl * 4 + mt) * 512 + lane * 16 + jj;
}

// read element (m,n) of a tile-packed G buffer (ISA 32-bit C/D 16x16 layout)
__device__ __forceinline__ float gload(const float* G, int m, int n) {
  int lane = (n & 15) + ((m & 8) ? 16 : 0);
  int v    = m & 7;
  return G[(size_t)((m >> 4) * NT + (n >> 4)) * 256 + v * 32 + lane];
}

__device__ __forceinline__ void grid_sync(unsigned* cnt, unsigned* gen) {
  __syncthreads();
  __threadfence();
  if (threadIdx.x == 0) {
    unsigned g = __hip_atomic_load(gen, __ATOMIC_RELAXED, __HIP_MEMORY_SCOPE_AGENT);
    unsigned a = __hip_atomic_fetch_add(cnt, 1u, __ATOMIC_ACQ_REL, __HIP_MEMORY_SCOPE_AGENT);
    if (a == (unsigned)(NBLK - 1)) {
      __hip_atomic_store(cnt, 0u, __ATOMIC_RELAXED, __HIP_MEMORY_SCOPE_AGENT);
      __hip_atomic_fetch_add(gen, 1u, __ATOMIC_ACQ_REL, __HIP_MEMORY_SCOPE_AGENT);
    } else {
      while (__hip_atomic_load(gen, __ATOMIC_ACQUIRE, __HIP_MEMORY_SCOPE_AGENT) == g)
        __builtin_amdgcn_s_sleep(1);
    }
  }
  __syncthreads();
}

#if GRU_ASYNC_LDS
__device__ __forceinline__ void wait_async8() {
#if __has_builtin(__builtin_amdgcn_s_wait_asynccnt)
  __builtin_amdgcn_s_wait_asynccnt(8);
#else
  asm volatile("s_wait_asynccnt 0x8" ::: "memory");
#endif
}
__device__ __forceinline__ void wait_async0() {
#if __has_builtin(__builtin_amdgcn_s_wait_asynccnt)
  __builtin_amdgcn_s_wait_asynccnt(0);
#else
  asm volatile("s_wait_asynccnt 0x0" ::: "memory");
#endif
}
// 32KB cooperative async copy: 256 threads x 8 x 16B, ASYNCcnt += 8 per wave.
__device__ __forceinline__ void issue_chunk_copy(const unsigned short* __restrict__ src,
                                                 unsigned short* dst) {
#pragma unroll
  for (int i = 0; i < 8; ++i) {
    int idx = (threadIdx.x + i * THREADS) * 8;   // ushort units, 16B each
    vi4_t* sg = (vi4_t*)(src + idx);             // drop const, generic bitcast
    vi4_t* dl = (vi4_t*)(dst + idx);
    __builtin_amdgcn_global_load_async_to_lds_b128(
        (as1_vi4*)sg, (as3_vi4*)dl, /*offset=*/0, /*cpol=*/0);
  }
}
#else
// Fallback: register-staged copy (loads clause together, then stores).
__device__ __forceinline__ void sync_chunk_copy(const unsigned short* __restrict__ src,
                                                unsigned short* dst) {
  const float4* s4 = (const float4*)src;
  float4* d4 = (float4*)dst;
  float4 tmp[8];
#pragma unroll
  for (int i = 0; i < 8; ++i) tmp[i] = s4[threadIdx.x + i * THREADS];
#pragma unroll
  for (int i = 0; i < 8; ++i) d4[threadIdx.x + i * THREADS] = tmp[i];
}
#endif

__device__ __forceinline__ const unsigned short* chunk_src(
    const unsigned short* __restrict__ xpk, const unsigned short* __restrict__ hp,
    int t, int kc) {
  return (kc < 4) ? xpk + (((size_t)t * 32 + kc * 8) * 4) * 512
                  : hp + ((size_t)(kc - 4) * 8 * 4) * 512;
}

// ---- setup kernels --------------------------------------------------------

__global__ void gru_init_sync(unsigned* cnt, unsigned* gen) {
  *cnt = 0u;
  *gen = 0u;
}

// cz[b][g] = (cj@w_ch.T + b_ch + he@w_zh.T + b_zh + b_ih)[b][g]   (f32, exact)
__global__ __launch_bounds__(THREADS)
void gru_context(const float* __restrict__ cj, const float* __restrict__ he,
                 const float* __restrict__ w_ch, const float* __restrict__ w_zh,
                 const float* __restrict__ b_ih, const float* __restrict__ b_ch,
                 const float* __restrict__ b_zh, float* __restrict__ cz) {
  int gid = blockIdx.x * THREADS + threadIdx.x;       // 64*3072 threads
  int b = gid / G3;
  int g = gid - b * G3;
  const float* cjr = cj + (size_t)b * HID;
  const float* her = he + (size_t)b * HID;
  const float* wc  = w_ch + (size_t)g * HID;
  const float* wz  = w_zh + (size_t)g * HID;
  float acc = b_ch[g] + b_zh[g] + b_ih[g];
#pragma unroll 4
  for (int k = 0; k < HID; ++k)
    acc += cjr[k] * wc[k] + her[k] * wz[k];
  cz[gid] = acc;
}

// Pack W = [w_ih | w_hh] viewed as B-matrix (K=2048, N=3072) into bf16
// WMMA B-fragments: frag (nt*64+kb), per-lane 16 contiguous bf16.
__global__ __launch_bounds__(THREADS)
void gru_pack_w(const float* __restrict__ w_ih, const float* __restrict__ w_hh,
                unsigned short* __restrict__ wpk) {
  int tid  = blockIdx.x * THREADS + threadIdx.x;      // NT*64*32 threads
  int lane = tid & 31;
  int frag = tid >> 5;                                // nt*64 + kb
  int kb   = frag & 63;
  int nt   = frag >> 6;
  int n    = nt * 16 + (lane & 15);
  int k0   = kb * 32 + ((lane & 16) ? 8 : 0);         // B layout: hi lanes K+8, jj>=8 K+16
  unsigned short* dst = wpk + (size_t)frag * 512 + lane * 16;
#pragma unroll
  for (int jj = 0; jj < 16; ++jj) {
    int K = k0 + jj + ((jj >= 8) ? 8 : 0);
    float v = (K < HID) ? w_ih[(size_t)n * HID + K]
                        : w_hh[(size_t)n * HID + (K - HID)];
    dst[jj] = f2bf(v);
  }
}

// Pack input_ (T,B,D) into bf16 WMMA A-fragments: frag ((t*32+kb)*4+mt).
__global__ __launch_bounds__(THREADS)
void gru_pack_x(const float* __restrict__ input_, unsigned short* __restrict__ xpk) {
  int tid  = blockIdx.x * THREADS + threadIdx.x;      // T*32*4*32 threads
  int lane = tid & 31;
  int frag = tid >> 5;                                // ((t*32+kb)*4+mt)
  int mt   = frag & 3;
  int kb   = (frag >> 2) & 31;
  int t    = frag >> 7;
  int m    = mt * 16 + (lane & 15);                   // batch row
  int k0   = kb * 32;
  const float* src = input_ + ((size_t)t * BATCH + m) * HID;
  unsigned short* dst = xpk + (size_t)frag * 512 + lane * 16;
#pragma unroll
  for (int jj = 0; jj < 16; ++jj) {
    int K;
    if (lane < 16) K = k0 + ((jj < 8) ? jj : (8 + jj));        // {0..7,16..23}
    else           K = k0 + ((jj < 8) ? (8 + jj) : (16 + jj)); // {8..15,24..31}
    dst[jj] = f2bf(src[K]);
  }
}

// h0 -> f32 state buffer slot 0 + packed bf16 A-fragments slot 0
__global__ __launch_bounds__(THREADS)
void gru_pack_h0(const float* __restrict__ h0, float* __restrict__ hbuf0,
                 unsigned short* __restrict__ hpk0) {
  int e = blockIdx.x * THREADS + threadIdx.x;         // B*H threads
  float v = h0[e];
  hbuf0[e] = v;
  hpk0[hpk_index(e >> 10, e & 1023)] = f2bf(v);
}

// ---- persistent recurrence kernel -----------------------------------------
// 48 blocks x 256 threads (8 waves). Block covers 4 n-tiles; each wave owns
// one n-tile half (2 m-tiles), i.e. 2 x-accumulators + 2 h-accumulators.
__global__ __launch_bounds__(THREADS)
void gru_recurrence(const unsigned short* __restrict__ xpk,
                    const unsigned short* __restrict__ wpk,
                    unsigned short* __restrict__ hpk,    // 2 ping-pong slots
                    float* __restrict__ hbuf,            // 2 ping-pong slots
                    const float* __restrict__ cz,
                    const float* __restrict__ b_hh,
                    const int*  __restrict__ length,
                    float* __restrict__ Gx,
                    float* __restrict__ Gh,
                    float* __restrict__ out,             // (T,B,H) then (B,H)
                    unsigned* __restrict__ sync_cnt,
                    unsigned* __restrict__ sync_gen) {
  const int lane = threadIdx.x & 31;
  const int wave = threadIdx.x >> 5;                   // 0..7
  const int nt   = blockIdx.x * 4 + (wave >> 1);       // 0..191
  const int mt0  = (wave & 1) * 2;                     // {0, 2}

  __shared__ unsigned short Abuf[2][CHUNK];            // 2 x 32 KB double buffer

  for (int t = 0; t < T_STEPS; ++t) {
    const unsigned short* hp = hpk + (size_t)(t & 1) * HPK_ELEMS;
    v8f ax0 = {}, ax1 = {}, ah0 = {}, ah1 = {};

#if GRU_ASYNC_LDS
    issue_chunk_copy(chunk_src(xpk, hp, t, 0), Abuf[0]);   // prime buffer 0
#endif

    // ---- step GEMM: [x_t | h] (64x2048 bf16) x W^T (2048x3072 bf16) ----
    for (int kc = 0; kc < 8; ++kc) {                   // 8 K-chunks of 256
      unsigned short* cur = Abuf[kc & 1];
#if GRU_ASYNC_LDS
      if (kc < 7) {
        issue_chunk_copy(chunk_src(xpk, hp, t, kc + 1), Abuf[(kc + 1) & 1]);
        wait_async8();                                 // chunk kc complete
      } else {
        wait_async0();
      }
#else
      sync_chunk_copy(chunk_src(xpk, hp, t, kc), cur);
#endif
      __syncthreads();                                 // cur visible to all waves

      const unsigned short* wbase =
          wpk + ((size_t)nt * 64 + kc * 8) * 512 + lane * 16;
      if (kc < 4) {                                    // x-projection part
#pragma unroll 2
        for (int k8 = 0; k8 < 8; ++k8) {
          v16bf bf = *(const v16bf*)(wbase + (size_t)k8 * 512);
          v16bf a0 = *(const v16bf*)(cur + (k8 * 4 + mt0) * 512 + lane * 16);
          v16bf a1 = *(const v16bf*)(cur + (k8 * 4 + mt0 + 1) * 512 + lane * 16);
          ax0 = bf16_mma(a0, bf, ax0);
          ax1 = bf16_mma(a1, bf, ax1);
        }
      } else {                                         // h-projection part
#pragma unroll 2
        for (int k8 = 0; k8 < 8; ++k8) {
          v16bf bf = *(const v16bf*)(wbase + (size_t)k8 * 512);
          v16bf a0 = *(const v16bf*)(cur + (k8 * 4 + mt0) * 512 + lane * 16);
          v16bf a1 = *(const v16bf*)(cur + (k8 * 4 + mt0 + 1) * 512 + lane * 16);
          ah0 = bf16_mma(a0, bf, ah0);
          ah1 = bf16_mma(a1, bf, ah1);
        }
      }
      __syncthreads();                                 // done with cur
    }

    // store tile-packed G buffers (per-lane element v at [v*32 + lane])
    {
      float* gx0 = Gx + (size_t)((mt0 + 0) * NT + nt) * 256;
      float* gx1 = Gx + (size_t)((mt0 + 1) * NT + nt) * 256;
      float* gh0 = Gh + (size_t)((mt0 + 0) * NT + nt) * 256;
      float* gh1 = Gh + (size_t)((mt0 + 1) * NT + nt) * 256;
#pragma unroll
      for (int v = 0; v < 8; ++v) {
        gx0[v * 32 + lane] = ax0[v];
        gx1[v * 32 + lane] = ax1[v];
        gh0[v * 32 + lane] = ah0[v];
        gh1[v * 32 + lane] = ah1[v];
      }
    }

    grid_sync(sync_cnt, sync_gen);

    // ---- gate phase: fused sigmoid/tanh, mask, output, h-repack ----
    const float* hprev = hbuf + (size_t)(t & 1) * (BATCH * HID);
    float* hnext = hbuf + (size_t)((t + 1) & 1) * (BATCH * HID);
    unsigned short* hpn = hpk + (size_t)((t + 1) & 1) * HPK_ELEMS;
    float* outt = out + (size_t)t * (BATCH * HID);

    for (int e = blockIdx.x * THREADS + threadIdx.x; e < BATCH * HID;
         e += NBLK * THREADS) {
      int b = e >> 10, j = e & 1023;
      const float* czb = cz + (size_t)b * G3;
      float gxr = gload(Gx, b, j),         ghr = gload(Gh, b, j);
      float gxi = gload(Gx, b, HID + j),   ghi = gload(Gh, b, HID + j);
      float gxn = gload(Gx, b, 2*HID + j), ghn = gload(Gh, b, 2*HID + j);
      float r  = sigmoid_f(gxr + ghr + czb[j]       + b_hh[j]);
      float ig = sigmoid_f(gxi + ghi + czb[HID + j] + b_hh[HID + j]);
      float ng = tanhf(gxn + czb[2*HID + j] + r * (ghn + b_hh[2*HID + j]));
      float hp_bj = hprev[e];
      float hy = ng + ig * (hp_bj - ng);
      // faithful to reference: masked branch broadcasts previous h row 0
      float hn = (t < length[b]) ? hy : hprev[j];
      hnext[e] = hn;
      outt[e]  = hn;
      hpn[hpk_index(b, j)] = f2bf(hn);
      if (t == T_STEPS - 1)
        out[(size_t)T_STEPS * BATCH * HID + e] = hn;   // h_n tail
    }

    grid_sync(sync_cnt, sync_gen);
  }
}

// ---- launcher -------------------------------------------------------------

extern "C" void kernel_launch(void* const* d_in, const int* in_sizes, int n_in,
                              void* d_out, int out_size, void* d_ws, size_t ws_size,
                              hipStream_t stream) {
  (void)in_sizes; (void)n_in; (void)out_size; (void)ws_size;
  const float* input_ = (const float*)d_in[0];
  const int*   length = (const int*)  d_in[1];
  const float* h0     = (const float*)d_in[2];
  const float* cj     = (const float*)d_in[3];
  const float* he     = (const float*)d_in[4];
  const float* w_ih   = (const float*)d_in[5];
  const float* w_hh   = (const float*)d_in[6];
  const float* w_ch   = (const float*)d_in[7];
  const float* w_zh   = (const float*)d_in[8];
  const float* b_ih   = (const float*)d_in[9];
  const float* b_hh   = (const float*)d_in[10];
  const float* b_ch   = (const float*)d_in[11];
  const float* b_zh   = (const float*)d_in[12];
  float* out = (float*)d_out;

  char* ws = (char*)d_ws;
  size_t off = 0;
  auto alloc = [&](size_t bytes) -> void* {
    void* p = ws + off;
    off += (bytes + 255) & ~(size_t)255;
    return p;
  };
  unsigned short* xpk = (unsigned short*)alloc((size_t)T_STEPS * 32 * 4 * 512 * 2); // 32 MB
  unsigned short* wpk = (unsigned short*)alloc((size_t)NT * 64 * 512 * 2);          // 12 MB
  unsigned short* hpk = (unsigned short*)alloc((size_t)2 * HPK_ELEMS * 2);
  float* hbuf = (float*)alloc((size_t)2 * BATCH * HID * 4);
  float* cz   = (float*)alloc((size_t)BATCH * G3 * 4);
  float* Gx   = (float*)alloc((size_t)4 * NT * 256 * 4);
  float* Gh   = (float*)alloc((size_t)4 * NT * 256 * 4);
  unsigned* syncw = (unsigned*)alloc(256);

  gru_init_sync<<<1, 1, 0, stream>>>(syncw, syncw + 1);
  gru_context<<<(BATCH * G3) / THREADS, THREADS, 0, stream>>>(
      cj, he, w_ch, w_zh, b_ih, b_ch, b_zh, cz);
  gru_pack_w<<<(NT * 64 * 32) / THREADS, THREADS, 0, stream>>>(w_ih, w_hh, wpk);
  gru_pack_x<<<(T_STEPS * 32 * 4 * 32) / THREADS, THREADS, 0, stream>>>(input_, xpk);
  gru_pack_h0<<<(BATCH * HID) / THREADS, THREADS, 0, stream>>>(h0, hbuf, hpk);
  gru_recurrence<<<NBLK, THREADS, 0, stream>>>(
      xpk, wpk, hpk, hbuf, cz, b_hh, length, Gx, Gh, out, syncw, syncw + 1);
}
